// GIN_35837207117956
// MI455X (gfx1250) — matrix operand order repaired and checked by reading, not compiled, not run on hardware
//
#include <hip/hip_runtime.h>

// GIN layer: B=16, N=1024, F=128, H=256
#define BN 16
#define NN 1024
#define FF 128
#define HH 256

typedef __attribute__((ext_vector_type(16))) _Float16 v16h;
typedef __attribute__((ext_vector_type(8)))  _Float16 v8h;
typedef __attribute__((ext_vector_type(8)))  float    v8f;
typedef unsigned int u32x4 __attribute__((ext_vector_type(4)));
typedef int          i32x4 __attribute__((ext_vector_type(4)));
typedef int          i32x8 __attribute__((ext_vector_type(8)));

#if defined(__has_builtin)
#if __has_builtin(__builtin_amdgcn_tensor_load_to_lds) && \
    __has_builtin(__builtin_amdgcn_s_wait_tensorcnt)
#define HAVE_TDM 1
#endif
#endif

// A-operand (16x32 f16, MxK) per-lane fragment, ISA 7.12.2:
//   lane<16 : row M=lane,    elems 0..7 -> K=0..7,  elems 8..15 -> K=16..23
//   lane>=16: row M=lane-16, elems 0..7 -> K=8..15, elems 8..15 -> K=24..31
static __device__ inline v16h make_a_frag(const _Float16* __restrict__ row, int kbase) {
  v8h lo = *(const v8h*)(row + kbase);
  v8h hi = *(const v8h*)(row + kbase + 16);
  v16h a;
#pragma unroll
  for (int i = 0; i < 8; ++i) { a[i] = lo[i]; a[i + 8] = hi[i]; }
  return a;
}

// B-operand (32x16 f16, KxN): lane<16 holds K=0..7,16..23 of col N=lane; lane>=16 K=8..15,24..31.
// col points at K-contiguous storage (LDS or global fp16); kb2 = 16*(lane>>4).
static __device__ inline v16h make_b_frag(const _Float16* __restrict__ col, int kb2) {
  v8h lo = *(const v8h*)(col + (kb2 >> 1));
  v8h hi = *(const v8h*)(col + (kb2 >> 1) + 16);
  v16h b;
#pragma unroll
  for (int i = 0; i < 8; ++i) { b[i] = lo[i]; b[i + 8] = hi[i]; }
  return b;
}

// pack two f32 -> two f16 in one dword
static __device__ inline unsigned pk2h(float a, float b) {
  union { _Float16 h[2]; unsigned u; } p;
  p.h[0] = (_Float16)a; p.h[1] = (_Float16)b;
  return p.u;
}

// ---------------------------------------------------------------------------
// Pre-pass: dst[z][c][r] = (f16)src[z][r][c]  (32x32 LDS tile transpose+cvt)
// grid = (R/32, C/32, batch), 256 threads
// ---------------------------------------------------------------------------
__global__ __launch_bounds__(256)
void transpose_cvt_kernel(const float* __restrict__ src,
                          _Float16* __restrict__ dst, int R, int C) {
  __shared__ _Float16 t[32][33];
  const int z = blockIdx.z;
  const float* s = src + (size_t)z * R * C;
  _Float16*    d = dst + (size_t)z * R * C;
  const int r0 = blockIdx.x * 32, c0 = blockIdx.y * 32;
  {
    int tr = threadIdx.x >> 3;          // 0..31 tile row
    int tc = (threadIdx.x & 7) * 4;     // 4 consecutive cols
    const float* p = &s[(size_t)(r0 + tr) * C + c0 + tc];
#pragma unroll
    for (int j = 0; j < 4; ++j) t[tc + j][tr] = (_Float16)p[j];
  }
  __syncthreads();
  {
    int oc  = threadIdx.x >> 3;         // dst row (= src col)
    int orr = (threadIdx.x & 7) * 4;    // 4 consecutive dst cols (= src rows)
    _Float16* p = &d[(size_t)(c0 + oc) * R + r0 + orr];
#pragma unroll
    for (int j = 0; j < 4; ++j) p[j] = t[oc][orr + j];
  }
}

// ---------------------------------------------------------------------------
// Kernel 1: x_agg = adj^T @ x, h = (1+eps)*x + x_agg -> fp16 ws.
// Only the adjacency transpose is LDS-staged (that is the HBM stream);
// B-fragments come K-contiguous from x16T[b][f][n] (L2-resident).
// ---------------------------------------------------------------------------
#define LAW 72   // lA row pitch (halves): 64 + pad, 144B = 16B multiple

__global__ __launch_bounds__(256)
void gin_aggregate_kernel(const float* __restrict__ x,
                          const _Float16* __restrict__ x16T,
                          const float* __restrict__ adj,
                          const float* __restrict__ eps,
                          _Float16* __restrict__ hbuf) {
  __shared__ _Float16 lA[16 * LAW];    // lA[m][k] = adj[b, n0+k, m0+m]

  const int tid  = threadIdx.x;
  const int lane = tid & 31;
  const int w    = tid >> 5;
  const int b    = blockIdx.x >> 6;
  const int m0   = (blockIdx.x & 63) << 4;

  const float epsv  = 1.0f + eps[0];
  const int   mrow  = lane & 15;
  const int   kbase = (lane >> 4) * 8;
  const int   kb2   = (lane >> 4) * 16;
  const int   f0    = w * 16;

  // this lane's B-fragment source row: x16T[b][f0+mrow][*]
  const _Float16* xrow = x16T + ((size_t)b * FF + f0 + mrow) * NN;

  v8f acc = {};

  for (int n0 = 0; n0 < NN; n0 += 64) {
    // stage adj^T 16m x 64k: 512 dword-pairs, packed b32 stores
#pragma unroll
    for (int rep = 0; rep < 2; ++rep) {
      int p = tid + rep * 256;
      int m = p & 15;
      int k = (p >> 4) * 2;
      const float* s = &adj[((size_t)b * NN + (n0 + k)) * NN + m0 + m];
      *(unsigned*)&lA[m * LAW + k] = pk2h(s[0], s[NN]);
    }
    if (n0 + 64 < NN)   // prefetch next adjacency slab (global_prefetch_b8)
      __builtin_prefetch(&adj[((size_t)b * NN + (n0 + 64 + lane)) * NN + m0], 0, 1);

    __syncthreads();
#pragma unroll
    for (int kk = 0; kk < 64; kk += 32) {
      v16h af = make_a_frag(&lA[mrow * LAW + kk], kbase);
      v16h bf = make_b_frag(xrow + n0 + kk, kb2);
      acc = __builtin_amdgcn_wmma_f32_16x16x32_f16(false, af, false, bf,
                                                   (short)0, acc, false, false);
    }
    __syncthreads();
  }

  // h = (1+eps)*x + x_agg ; D layout: row = r + 8*(lane>>4), col = lane&15
#pragma unroll
  for (int r = 0; r < 8; ++r) {
    int    m   = m0 + r + ((lane >> 4) << 3);
    size_t idx = ((size_t)b * NN + m) * FF + f0 + mrow;
    hbuf[idx] = (_Float16)(epsv * x[idx] + acc[r]);
  }
}

// ---------------------------------------------------------------------------
// Kernel 2: y = relu(h@W1 + b1)@W2 + b2, masked. h tile staged by the Tensor
// Data Mover; weight B-fragments load K-contiguous from fp16 W1T/W2T (L2).
// No barriers inside the K loops.
// ---------------------------------------------------------------------------
#define LHW   136                       // lH pitch: 128 + 8 pad = 64+4 dwords
#define LACTW 264                       // lAct pitch
#define SMEM2_BYTES ((16 * LHW + 16 * LACTW) * 2)

__global__ __launch_bounds__(256)
void gin_mlp_kernel(const _Float16* __restrict__ hbuf,
                    const _Float16* __restrict__ W1T,   // [HH][FF] = W1^T
                    const float* __restrict__ b1,
                    const _Float16* __restrict__ W2T,   // [FF][HH] = W2^T
                    const float* __restrict__ b2,
                    const unsigned char* __restrict__ mask,
                    float* __restrict__ out) {
  extern __shared__ _Float16 smem[];
  _Float16* lH   = smem;                 // [16][LHW] (offset 0 for TDM)
  _Float16* lAct = smem + 16 * LHW;      // [16][LACTW]

  const int tid  = threadIdx.x;
  const int lane = tid & 31;
  const int w    = tid >> 5;
  const int b    = blockIdx.x >> 6;
  const int m0   = (blockIdx.x & 63) << 4;

  const int mrow  = lane & 15;
  const int kbase = (lane >> 4) * 8;
  const int kb2   = (lane >> 4) * 16;

  // ---- stage h tile [16][128] fp16 into lH via TDM (one wave issues DMA)
#if HAVE_TDM
  if (w == 0) {
    unsigned long long gaddr =
        (unsigned long long)(const void*)(hbuf + ((size_t)b * NN + m0) * FF);
    u32x4 g0;
    g0[0] = 1u;                                   // count=1, user descriptor
    g0[1] = 0u;                                   // lds_addr = offset of lH
    g0[2] = (unsigned)(gaddr & 0xffffffffu);
    g0[3] = (unsigned)((gaddr >> 32) & 0x1ffffffu) | (2u << 30);  // type=2
    i32x8 g1;
    g1[0] = (1 << 16)      // data_size = 2 bytes
          | (1 << 20)      // pad_enable
          | (5 << 22)      // pad_interval: every 64 dwords (=128 halves)
          | (3 << 25);     // pad_amount: 4 dwords (=8 halves) -> pitch 136
    g1[1] = (int)(128u << 16);   // tensor_dim0 = 128  (bits 79:48)
    g1[2] = (int)(16u << 16);    // tensor_dim1 = 16   (bits 111:80)
    g1[3] = (int)(128u << 16);   // tile_dim0  = 128   (bits 127:112)
    g1[4] = 16;                  // tile_dim1  = 16    (bits 143:128)
    g1[5] = 128;                 // tensor_dim0_stride (bits 207:160)
    g1[6] = 0;
    g1[7] = 0;
    i32x4 z4 = {0, 0, 0, 0};
#if __clang_major__ >= 23
    i32x8 z8 = {0, 0, 0, 0, 0, 0, 0, 0};
    __builtin_amdgcn_tensor_load_to_lds(g0, g1, z4, z4, z8, 0);
#else
    __builtin_amdgcn_tensor_load_to_lds(g0, g1, z4, z4, 0);
#endif
    __builtin_amdgcn_s_wait_tensorcnt(0);
  }
#else
  {
    int i = tid * 8, m = i >> 7, f = i & 127;
    const _Float16* src = &hbuf[((size_t)b * NN + m0 + m) * FF + f];
#pragma unroll
    for (int j = 0; j < 8; ++j) lH[m * LHW + f + j] = src[j];
  }
#endif
  __syncthreads();

  // ---- stage 1: act = relu(h @ W1 + b1); wave w owns columns [32w, 32w+32)
  const int c0 = (2 * w + 0) * 16 + mrow;
  const int c1 = (2 * w + 1) * 16 + mrow;
  v8f acc0 = {}, acc1 = {};
#pragma unroll
  for (int k0 = 0; k0 < FF; k0 += 32) {
    v16h af  = make_a_frag(lH + mrow * LHW + k0, kbase);
    v16h bf0 = make_b_frag(W1T + (size_t)c0 * FF + k0, kb2);
    v16h bf1 = make_b_frag(W1T + (size_t)c1 * FF + k0, kb2);
    acc0 = __builtin_amdgcn_wmma_f32_16x16x32_f16(false, af, false, bf0,
                                                  (short)0, acc0, false, false);
    acc1 = __builtin_amdgcn_wmma_f32_16x16x32_f16(false, af, false, bf1,
                                                  (short)0, acc1, false, false);
  }
#pragma unroll
  for (int r = 0; r < 8; ++r) {
    int m = r + ((lane >> 4) << 3);
    lAct[m * LACTW + c0] = (_Float16)fmaxf(acc0[r] + b1[c0], 0.0f);
    lAct[m * LACTW + c1] = (_Float16)fmaxf(acc1[r] + b1[c1], 0.0f);
  }
  __syncthreads();

  // ---- stage 2: y = act @ W2 + b2; wave w owns feature tile w*16
  const int f = w * 16 + mrow;
  v8f acc2 = {};
#pragma unroll
  for (int k0 = 0; k0 < HH; k0 += 32) {
    v16h af = make_a_frag(lAct + mrow * LACTW + k0, kbase);
    v16h bf = make_b_frag(W2T + (size_t)f * HH + k0, kb2);
    acc2 = __builtin_amdgcn_wmma_f32_16x16x32_f16(false, af, false, bf,
                                                  (short)0, acc2, false, false);
  }
#pragma unroll
  for (int r = 0; r < 8; ++r) {
    int  m  = m0 + r + ((lane >> 4) << 3);
    bool mk = mask[(size_t)b * NN + m] != 0;
    float y = acc2[r] + b2[f];
    out[((size_t)b * NN + m) * FF + f] = mk ? y : 0.0f;
  }
}

// ---------------------------------------------------------------------------
extern "C" void kernel_launch(void* const* d_in, const int* in_sizes, int n_in,
                              void* d_out, int out_size, void* d_ws, size_t ws_size,
                              hipStream_t stream) {
  (void)in_sizes; (void)n_in; (void)out_size; (void)ws_size;
  const float*         x    = (const float*)d_in[0];
  const float*         adj  = (const float*)d_in[1];
  const unsigned char* mask = (const unsigned char*)d_in[2];  // jax bool = 1 byte
  const float*         W1   = (const float*)d_in[3];
  const float*         b1   = (const float*)d_in[4];
  const float*         W2   = (const float*)d_in[5];
  const float*         b2   = (const float*)d_in[6];
  const float*         eps  = (const float*)d_in[7];

  // workspace layout (all 256B aligned)
  char*     ws   = (char*)d_ws;
  _Float16* hbuf = (_Float16*)ws;                               // 4 MiB
  _Float16* x16T = (_Float16*)(ws + (size_t)4 * 1024 * 1024);   // 4 MiB
  _Float16* W1T  = (_Float16*)(ws + (size_t)8 * 1024 * 1024);   // 64 KiB
  _Float16* W2T  = (_Float16*)(ws + (size_t)8 * 1024 * 1024 + 65536);  // 64 KiB
  float*    out  = (float*)d_out;

  // pre-pass: fp16 transposed copies (one-time, L2-resident afterwards)
  transpose_cvt_kernel<<<dim3(NN / 32, FF / 32, BN), 256, 0, stream>>>(x, x16T, NN, FF);
  transpose_cvt_kernel<<<dim3(FF / 32, HH / 32, 1), 256, 0, stream>>>(W1, W1T, FF, HH);
  transpose_cvt_kernel<<<dim3(HH / 32, FF / 32, 1), 256, 0, stream>>>(W2, W2T, HH, FF);

  dim3 grid(BN * (NN / 16));  // 1024 blocks
  dim3 block(256);            // 8 wave32s
  gin_aggregate_kernel<<<grid, block, 0, stream>>>(x, x16T, adj, eps, hbuf);
  gin_mlp_kernel<<<grid, block, SMEM2_BYTES, stream>>>(hbuf, W1T, b1, W2T, b2,
                                                       mask, out);
}
